// ViT_33105607918191
// MI455X (gfx1250) — matrix operand order repaired and checked by reading, not compiled, not run on hardware
//
#include <hip/hip_runtime.h>
#include <hip/hip_bf16.h>
#include <math.h>

typedef __attribute__((ext_vector_type(16))) __bf16 v16bf;
typedef __attribute__((ext_vector_type(8)))  float  v8f;

#define B_    8
#define T_    4
#define K_    128
#define H_    64
#define W_    64
#define HW_   4096
#define KHW_  (K_*HW_)
#define TINV  (1.0f/0.07f)
#define NERR  98304            // B*3*HW
#define KDROP 39321            // int(0.4*NERR)
#define NEEDED (NERR-KDROP)    // 58983
#define KRANK (KDROP+1)        // 1-based rank of smallest kept value
#define NB    96

// ---- monotone float<->uint key (ascending) ----
__device__ inline unsigned mono(float f){
  unsigned u = __float_as_uint(f);
  return (u & 0x80000000u) ? ~u : (u | 0x80000000u);
}
__device__ inline float unmono(unsigned k){
  unsigned u = (k & 0x80000000u) ? (k & 0x7fffffffu) : ~k;
  return __uint_as_float(u);
}

// nearest-neighbor affine warp source pixel; -1 if invalid (==> zero features)
__device__ inline int warp_src(const float* th, int h, int w){
  float x = (2*w + 1) * (1.0f/W_) - 1.0f;
  float y = (2*h + 1) * (1.0f/H_) - 1.0f;
  float gx = th[0]*x + th[1]*y + th[2];
  float gy = th[3]*x + th[4]*y + th[5];
  float fx = rintf(((gx + 1.0f)*W_ - 1.0f)*0.5f);   // round-half-even like jnp.round
  float fy = rintf(((gy + 1.0f)*H_ - 1.0f)*0.5f);
  if(!(fx >= 0.0f && fx < (float)W_ && fy >= 0.0f && fy < (float)H_)) return -1;
  return (int)fy * W_ + (int)fx;
}

// ---------------- 1) build 128x128 bf16 query matrix ----------------
__global__ void build_queries(const float* __restrict__ key1, const int* __restrict__ ox1,
                              const int* __restrict__ oy1, __bf16* __restrict__ Q){
  int idx = blockIdx.x*blockDim.x + threadIdx.x;
  if(idx >= 128*128) return;
  int i = idx >> 7, c = idx & 127;
  int b = i >> 4, q = i & 15, qx = q & 3, qy = q >> 2;
  int pos = (qy*16 + oy1[b])*W_ + (qx*16 + ox1[b]);
  Q[idx] = (__bf16)key1[(b*(T_+1))*KHW_ + c*HW_ + pos];
}

// ---------------- 2) temporal loss: fused gather + WMMA GEMM + CE ----------------
__global__ __launch_bounds__(128) void temporal_loss(
    const float* __restrict__ key1, const float* __restrict__ key2raw,
    const float* __restrict__ affines, const float* __restrict__ affine_src,
    const __bf16* __restrict__ Qg, float* __restrict__ errOut)
{
  __shared__ __bf16 Qs[128][128];       // 32 KB, shared by WG
  __shared__ __bf16 Abuf[4][16][128];   // 16 KB, per-wave gathered A tile
  __shared__ int    pidx[4][16];
  __shared__ int    spos[4][2][16];

  const int frame = blockIdx.x >> 6;    // 0..23
  const int blk   = blockIdx.x & 63;
  const int b = frame / 3;
  const int t = frame % 3 + 1;
  const int wave = threadIdx.x >> 5;
  const int lane = threadIdx.x & 31;
  const int pixBase = blk*64 + wave*16;
  const int rm    = lane & 15;          // A row / B column within tile
  const int khalf = lane >> 4;

  // stage Q into LDS (dword copies)
  {
    const unsigned* qsrc = (const unsigned*)Qg;
    unsigned* qdst = (unsigned*)&Qs[0][0];
    for(int i = threadIdx.x; i < 128*128/2; i += 128) qdst[i] = qsrc[i];
  }
  // warp source pixels for this wave's 16 output pixels
  if(lane < 16){
    int p = pixBase + lane, h = p >> 6, w = p & 63;
    spos[wave][0][lane] = warp_src(affines    + (b*T_ + t)*6, h, w);
    spos[wave][1][lane] = warp_src(affine_src + (b*T_ + t)*6, h, w);
  }
  __syncthreads();

  // ---------- pseudo phase: gather k2 tile, 16x16x128 GEMM, windowed argmax ----------
  {
    const float* f2 = key2raw + (b*(T_-1) + (t-1))*KHW_;
    for(int i = lane; i < 2048; i += 32){
      int m = i >> 7, c = i & 127;
      int pos = spos[wave][1][m];
      Abuf[wave][m][c] = (__bf16)((pos >= 0) ? f2[c*HW_ + pos] : 0.0f);
    }
  }
  __syncthreads();

  v8f accP; 
  #pragma unroll
  for(int r = 0; r < 8; ++r) accP[r] = 0.0f;

  #pragma unroll
  for(int s = 0; s < 4; ++s){
    v16bf af, bfm;
    #pragma unroll
    for(int e = 0; e < 16; ++e){
      int v  = e >> 1;
      int kl = 16*khalf + e;   // A: K=16*khalf+e (matches B)
      af[e]  = Abuf[wave][rm][32*s + kl];
      bfm[e] = Qs[16*b + rm][32*s + 16*khalf + e];                  // B: K=16*khalf+e, col=rm
    }
    accP = __builtin_amdgcn_wmma_f32_16x16x32_bf16(false, af, false, bfm,
                                                   (short)0, accP, false, false);
  }
  #pragma unroll
  for(int r = 0; r < 8; ++r){
    float v = accP[r]; int c = rm;
    #pragma unroll
    for(int m = 1; m < 16; m <<= 1){
      float ov = __shfl_xor(v, m, 32);
      int   oc = __shfl_xor(c, m, 32);
      if(ov > v || (ov == v && oc < c)){ v = ov; c = oc; }          // first-index tie-break
    }
    if(rm == 0) pidx[wave][r + 8*khalf] = 16*b + c;
  }
  __syncthreads();

  // ---------- soft phase: gather k1 tile, 16x128x128 GEMM, log-softmax CE ----------
  {
    const float* f1 = key1 + (b*(T_+1) + t)*KHW_;
    for(int i = lane; i < 2048; i += 32){
      int m = i >> 7, c = i & 127;
      int pos = spos[wave][0][m];
      Abuf[wave][m][c] = (__bf16)((pos >= 0) ? f1[c*HW_ + pos] : 0.0f);
    }
  }
  __syncthreads();

  v8f acc[8];
  #pragma unroll
  for(int tl = 0; tl < 8; ++tl)
    #pragma unroll
    for(int r = 0; r < 8; ++r) acc[tl][r] = 0.0f;

  #pragma unroll
  for(int s = 0; s < 4; ++s){
    v16bf af;
    #pragma unroll
    for(int e = 0; e < 16; ++e){
      int v  = e >> 1;
      int kl = 16*khalf + e;
      af[e]  = Abuf[wave][rm][32*s + kl];
    }
    #pragma unroll
    for(int tl = 0; tl < 8; ++tl){
      v16bf bfm;
      #pragma unroll
      for(int e = 0; e < 16; ++e)
        bfm[e] = Qs[16*tl + rm][32*s + 16*khalf + e];
      acc[tl] = __builtin_amdgcn_wmma_f32_16x16x32_bf16(false, af, false, bfm,
                                                        (short)0, acc[tl], false, false);
    }
  }

  #pragma unroll
  for(int r = 0; r < 8; ++r){
    int row = r + 8*khalf;
    float mx = acc[0][r];
    #pragma unroll
    for(int tl = 1; tl < 8; ++tl) mx = fmaxf(mx, acc[tl][r]);
    #pragma unroll
    for(int m = 1; m < 16; m <<= 1) mx = fmaxf(mx, __shfl_xor(mx, m, 32));

    int   pc = pidx[wave][row];
    float se = 0.0f, sv = 0.0f;
    #pragma unroll
    for(int tl = 0; tl < 8; ++tl){
      se += expf((acc[tl][r] - mx) * TINV);
      if(16*tl + rm == pc) sv = acc[tl][r];
    }
    #pragma unroll
    for(int m = 1; m < 16; m <<= 1){
      se += __shfl_xor(se, m, 32);
      sv += __shfl_xor(sv, m, 32);
    }
    float errv = (mx - sv) * TINV + logf(se);
    if(rm == 0) errOut[frame*HW_ + pixBase + row] = errv;
  }
}

// ---------------- 3) reference branch: gather x1/y1 ----------------
__global__ void ref_gather(const float* __restrict__ key1,
                           const float* __restrict__ affines, const float* __restrict__ affine_src,
                           const int* __restrict__ ox2, const int* __restrict__ oy2,
                           float* __restrict__ x1, float* __restrict__ y1){
  int i = blockIdx.x, c = threadIdx.x;           // 128 x 128
  int b = i >> 4, q = i & 15, qx = q & 3, qy = q >> 2;
  int w = qx*16 + ox2[b], h = qy*16 + oy2[b];
  int ps = warp_src(affines    + (b*T_)*6, h, w);
  int pp = warp_src(affine_src + (b*T_)*6, h, w);
  const float* fr1 = key1 + (b*(T_+1) + 0 )*KHW_;   // k1[b,0]
  const float* fr2 = key1 + (b*(T_+1) + T_)*KHW_;   // k2[b,0] == key1[b,4]
  x1[i*128 + c] = (ps >= 0) ? fr1[c*HW_ + ps] : 0.0f;
  y1[i*128 + c] = (pp >= 0) ? fr2[c*HW_ + pp] : 0.0f;
}

// 128x128x128 logits + diag CE (tiny; scalar f32)
__global__ void ref_logits(const float* __restrict__ x1, const float* __restrict__ y1,
                           float* __restrict__ lref){
  int i = threadIdx.x;
  float row[128];
  for(int j = 0; j < 128; ++j){
    float d = 0.0f;
    #pragma unroll 4
    for(int c = 0; c < 128; ++c) d += x1[i*128 + c] * y1[j*128 + c];
    row[j] = d * TINV;
  }
  float mx = row[0];
  for(int j = 1; j < 128; ++j) mx = fmaxf(mx, row[j]);
  float se = 0.0f;
  for(int j = 0; j < 128; ++j) se += expf(row[j] - mx);
  lref[i] = (mx - row[i]) + logf(se);
}

// sort 128 values (even-odd), mean of sorted[51:128]
__global__ void ref_sort(const float* __restrict__ lref, float* __restrict__ scal){
  __shared__ float s[128];
  int i = threadIdx.x;
  s[i] = lref[i];
  __syncthreads();
  for(int ph = 0; ph < 128; ++ph){
    int idx = 2*i + (ph & 1);
    if(i < 64 && idx + 1 < 128){
      float a = s[idx], b = s[idx+1];
      if(a > b){ s[idx] = b; s[idx+1] = a; }
    }
    __syncthreads();
  }
  if(i == 0){
    float sum = 0.0f;
    for(int k = 51; k < 128; ++k) sum += s[k];
    scal[0] = sum / 77.0f;
  }
}

// ---------------- 4) OHEM on err: exact 2x16-bit radix select ----------------
__global__ void ws_init(unsigned* hist, float* ps, unsigned* pc){
  int i = blockIdx.x*blockDim.x + threadIdx.x;
  if(i < 65536) hist[i] = 0u;
  if(i < NB){ ps[i] = 0.0f; pc[i] = 0u; }
}

__global__ void hist_hi(const float* __restrict__ err, unsigned* __restrict__ hist){
  for(int i = blockIdx.x*blockDim.x + threadIdx.x; i < NERR; i += gridDim.x*blockDim.x)
    atomicAdd(&hist[mono(err[i]) >> 16], 1u);
}

__global__ void select_hi(unsigned* __restrict__ hist, unsigned* __restrict__ sel){
  __shared__ unsigned cs[256];
  int t = threadIdx.x;
  unsigned sum = 0;
  for(int j = 0; j < 256; ++j) sum += hist[t*256 + j];
  cs[t] = sum;
  __syncthreads();
  if(t == 0){
    unsigned cum = 0; int ch = 0;
    while(ch < 255 && cum + cs[ch] < KRANK){ cum += cs[ch]; ++ch; }
    unsigned cum2 = cum; int binsel = ch*256;
    for(int j = 0; j < 256; ++j){
      unsigned h = hist[ch*256 + j];
      if(cum2 + h >= KRANK){ binsel = ch*256 + j; break; }
      cum2 += h;
    }
    sel[0] = (unsigned)binsel;  // hi bin of k-th smallest key
    sel[1] = cum2;              // #keys strictly below that hi bin region
  }
  __syncthreads();
  for(int j = t; j < 65536; j += 256) hist[j] = 0u;   // reset for pass 2
}

__global__ void hist_lo(const float* __restrict__ err, unsigned* __restrict__ hist,
                        const unsigned* __restrict__ sel){
  unsigned hi = sel[0];
  for(int i = blockIdx.x*blockDim.x + threadIdx.x; i < NERR; i += gridDim.x*blockDim.x){
    unsigned k = mono(err[i]);
    if((k >> 16) == hi) atomicAdd(&hist[k & 0xFFFFu], 1u);
  }
}

__global__ void select_lo(const unsigned* __restrict__ hist, unsigned* __restrict__ sel){
  __shared__ unsigned cs[256];
  int t = threadIdx.x;
  unsigned sum = 0;
  for(int j = 0; j < 256; ++j) sum += hist[t*256 + j];
  cs[t] = sum;
  __syncthreads();
  if(t == 0){
    unsigned kr = (unsigned)KRANK - sel[1];
    unsigned cum = 0; int ch = 0;
    while(ch < 255 && cum + cs[ch] < kr){ cum += cs[ch]; ++ch; }
    int lo = ch*256;
    for(int j = 0; j < 256; ++j){
      unsigned h = hist[ch*256 + j];
      if(cum + h >= kr){ lo = ch*256 + j; break; }
      cum += h;
    }
    sel[2] = (sel[0] << 16) | (unsigned)lo;   // exact k-th smallest key
  }
}

// deterministic partial sums of values strictly greater than key*
__global__ void sum_top(const float* __restrict__ err, const unsigned* __restrict__ sel,
                        float* __restrict__ ps, unsigned* __restrict__ pc){
  __shared__ float    sdata[256];
  __shared__ unsigned scnt[256];
  int tid = threadIdx.x;
  unsigned kstar = sel[2];
  float s = 0.0f; unsigned c = 0;
  for(int i = blockIdx.x*256 + tid; i < NERR; i += NB*256){
    float v = err[i];
    if(mono(v) > kstar){ s += v; ++c; }
  }
  sdata[tid] = s; scnt[tid] = c;
  __syncthreads();
  for(int st = 128; st > 0; st >>= 1){
    if(tid < st){ sdata[tid] += sdata[tid + st]; scnt[tid] += scnt[tid + st]; }
    __syncthreads();
  }
  if(tid == 0){ ps[blockIdx.x] = sdata[0]; pc[blockIdx.x] = scnt[0]; }
}

__global__ void final_combine(const float* __restrict__ ps, const unsigned* __restrict__ pc,
                              const unsigned* __restrict__ sel, const float* __restrict__ scal,
                              float* __restrict__ out){
  float S = 0.0f; unsigned C = 0;
  for(int i = 0; i < NB; ++i){ S += ps[i]; C += pc[i]; }
  float vstar = unmono(sel[2]);
  S += (float)(NEEDED - C) * vstar;           // fill with copies of the threshold value (ties)
  out[0] = 0.1f * scal[0] + S / (float)NEEDED;
}

// ---------------- launch ----------------
extern "C" void kernel_launch(void* const* d_in, const int* in_sizes, int n_in,
                              void* d_out, int out_size, void* d_ws, size_t ws_size,
                              hipStream_t stream){
  const float* key1       = (const float*)d_in[0];
  const float* key2raw    = (const float*)d_in[1];
  const float* affines    = (const float*)d_in[2];
  const float* affine_src = (const float*)d_in[3];
  const int*   ox1 = (const int*)d_in[4];
  const int*   oy1 = (const int*)d_in[5];
  const int*   ox2 = (const int*)d_in[6];
  const int*   oy2 = (const int*)d_in[7];

  char* ws = (char*)d_ws;
  __bf16*   Q    = (__bf16*)  (ws + 0);        // 32768 B
  float*    err  = (float*)   (ws + 32768);    // 393216 B
  float*    x1   = (float*)   (ws + 425984);   // 65536 B
  float*    y1   = (float*)   (ws + 491520);   // 65536 B
  float*    lref = (float*)   (ws + 557056);   // 512 B
  float*    scal = (float*)   (ws + 557568);   // 64 B
  unsigned* sel  = (unsigned*)(ws + 557632);   // 64 B
  unsigned* hist = (unsigned*)(ws + 557696);   // 262144 B
  float*    ps   = (float*)   (ws + 819840);   // 384 B
  unsigned* pc   = (unsigned*)(ws + 820224);   // 384 B

  build_queries<<<64, 256, 0, stream>>>(key1, ox1, oy1, Q);
  ref_gather<<<128, 128, 0, stream>>>(key1, affines, affine_src, ox2, oy2, x1, y1);
  temporal_loss<<<24*64, 128, 0, stream>>>(key1, key2raw, affines, affine_src, Q, err);
  ref_logits<<<1, 128, 0, stream>>>(x1, y1, lref);
  ref_sort<<<1, 128, 0, stream>>>(lref, scal);
  ws_init<<<256, 256, 0, stream>>>(hist, ps, pc);
  hist_hi<<<NB, 256, 0, stream>>>(err, hist);
  select_hi<<<1, 256, 0, stream>>>(hist, sel);
  hist_lo<<<NB, 256, 0, stream>>>(err, hist, sel);
  select_lo<<<1, 256, 0, stream>>>(hist, sel);
  sum_top<<<NB, 256, 0, stream>>>(err, sel, ps, pc);
  final_combine<<<1, 1, 0, stream>>>(ps, pc, sel, scal, (float*)d_out);
}
